// MemoryEfficientLinearCEModule_71674414235708
// MI455X (gfx1250) — matrix operand order repaired and checked by reading, not compiled, not run on hardware
//
#include <hip/hip_runtime.h>
#include <hip/hip_bf16.h>

typedef _Float16 v16h __attribute__((ext_vector_type(16)));
typedef _Float16 v8h  __attribute__((ext_vector_type(8)));
typedef _Float16 v4h  __attribute__((ext_vector_type(4)));
typedef float    v8f  __attribute__((ext_vector_type(8)));
typedef float    v4f  __attribute__((ext_vector_type(4)));

#define M_ROWS  4096
#define K_DIM   1024
#define N_COLS  128000
#define BM      128            // rows per block (8 row-tiles of 16)
#define NT      2              // col tiles per wave
#define BN      256            // cols per group (8 waves * NT * 16)
#define NGRP    4              // col groups per block -> 1024 cols/block
#define NCHUNK  (N_COLS / (BN * NGRP))   // 125 partial-lse chunks
#define KB      64             // K block through LDS
#define KSTEPS  (KB / 32)      // 2 wmma k-steps per K block
#define NKB     (K_DIM / KB)   // 16
#define NSTEPS  (NGRP * NKB)   // 64 staging steps per block
#define LDS_STR (KB + 8)       // 72 halfs: padded row -> conflict-free b128 reads
#define IGNORE_IDX (-100)

// sched_group_barrier masks (AMDGPU): 0x8 = MFMA/WMMA, 0x100 = DS read
#if __has_builtin(__builtin_amdgcn_sched_group_barrier)
#define SGB(mask, size) __builtin_amdgcn_sched_group_barrier((mask), (size), 0)
#else
#define SGB(mask, size)
#endif

// LDS-only workgroup barrier: orders DS ops across the workgroup WITHOUT
// draining loadcnt (keeps the global prefetch pipeline in flight).
__device__ __forceinline__ void wg_barrier_lds() {
    asm volatile(
        "s_wait_dscnt 0x0\n\t"
        "s_barrier_signal -1\n\t"
        "s_barrier_wait -1"
        ::: "memory");
}

__device__ __forceinline__ v16h load_frag16(const _Float16* p, int hi_off) {
    v8h lo = *(const v8h*)p;
    v8h hi = *(const v8h*)(p + hi_off);
    return __builtin_shufflevector(lo, hi, 0,1,2,3,4,5,6,7,8,9,10,11,12,13,14,15);
}

// ---------------- main fused GEMM + online-softmax kernel ----------------
__launch_bounds__(256)
__global__ void ce_main_kernel(const float* __restrict__ x,
                               const float* __restrict__ w,
                               const float* __restrict__ bias,
                               const long long* __restrict__ tgt,
                               float* __restrict__ pmax,
                               float* __restrict__ psum,
                               float* __restrict__ ptgt)
{
    __shared__ __align__(16) _Float16 Alds[BM * LDS_STR];   // 18432 B
    __shared__ __align__(16) _Float16 Blds[BN * LDS_STR];   // 36864 B
    __shared__ float wmx[8][BM];                            // 4096 B
    __shared__ float wsm[8][BM];                            // 4096 B
    __shared__ int   tl[BM];                                // 512 B

    const int tid   = threadIdx.x;
    const int wave  = tid >> 5;
    const int lane  = tid & 31;
    const int l16   = lane & 15;
    const int g     = lane >> 4;
    const int chunk = blockIdx.x;          // 0..NCHUNK-1
    const int m0    = blockIdx.y * BM;     // global row base
    const int nblk  = chunk * (BN * NGRP); // first column of this block

    // staging decomposition: tid&15 invariant, rows stride 16 per chunk-load
    const int tr  = tid >> 4;              // base row within slice
    const int c4  = (tid & 15) << 2;       // half/float index within row

    if (tid < BM) tl[tid] = (int)tgt[m0 + tid];   // low 32 bits: range fits, -100 preserved

    float run_m = -__builtin_inff();
    float run_s = 0.f;

    // register staging buffers (global prefetch pipeline)
    v4f pa[8];
    v4f pb[16];

    // issue global loads for staging step s = ng*NKB + kb (always in-bounds)
    auto issue_loads = [&](int s) {
        const int ng = s / NKB;
        const int k0 = (s % NKB) * KB;
        const int nb = nblk + ng * BN;
        #pragma unroll
        for (int j = 0; j < 8; ++j)
            pa[j] = *(const v4f*)&x[(size_t)(m0 + tr + j * 16) * K_DIM + k0 + c4];
        #pragma unroll
        for (int j = 0; j < 16; ++j)
            pb[j] = *(const v4f*)&w[(size_t)(nb + tr + j * 16) * K_DIM + k0 + c4];
    };
    // convert staged registers to f16 and commit to LDS
    auto commit_stage = [&]() {
        #pragma unroll
        for (int j = 0; j < 8; ++j) {
            v4h ah;
            #pragma unroll
            for (int i = 0; i < 4; ++i) ah[i] = (_Float16)pa[j][i];
            *(v4h*)&Alds[(tr + j * 16) * LDS_STR + c4] = ah;
        }
        #pragma unroll
        for (int j = 0; j < 16; ++j) {
            v4h bh;
            #pragma unroll
            for (int i = 0; i < 4; ++i) bh[i] = (_Float16)pb[j][i];
            *(v4h*)&Blds[(tr + j * 16) * LDS_STR + c4] = bh;
        }
    };
    // A fragment for flat step i: ks = i>>3, rt = i&7
    auto load_afrag = [&](int i) -> v16h {
        const int ks = i >> 3;
        const int rt = i & 7;
        return load_frag16(&Alds[(rt * 16 + l16) * LDS_STR + ks * 32 + g * 8], 16);
    };

    issue_loads(0);   // prologue prefetch

    for (int ng = 0; ng < NGRP; ++ng) {
        const int nbase = nblk + ng * BN;

        v8f acc[8][NT];
        #pragma unroll
        for (int rt = 0; rt < 8; ++rt)
            #pragma unroll
            for (int nt = 0; nt < NT; ++nt)
                #pragma unroll
                for (int v = 0; v < 8; ++v) acc[rt][nt][v] = 0.f;

        for (int kb = 0; kb < NKB; ++kb) {
            wg_barrier_lds();         // previous tile compute done reading LDS
            commit_stage();           // waits on this step's global loads, cvt + ds_store
            const int snext = ng * NKB + kb + 1;
            issue_loads(snext < NSTEPS ? snext : 0);  // stays in flight across barrier
            wg_barrier_lds();

            // preload B fragments for BOTH k-steps (reused by all 8 row tiles)
            v16h bf[KSTEPS][NT];
            #pragma unroll
            for (int ks = 0; ks < KSTEPS; ++ks)
                #pragma unroll
                for (int nt = 0; nt < NT; ++nt)
                    bf[ks][nt] = load_frag16(
                        &Blds[(wave * (NT * 16) + nt * 16 + l16) * LDS_STR + ks * 32 + g * 8], 16);

            // flat (ks, rt) loop with 4-buffer / 3-ahead A-fragment rotation
            v16h af[4];
            af[0] = load_afrag(0);
            af[1] = load_afrag(1);
            af[2] = load_afrag(2);
            // pin pipeline shape: prologue = 8 bf + 6 af DS reads
            SGB(0x100, 14);
            #pragma unroll
            for (int i = 0; i < KSTEPS * 8; ++i) {
                if (i + 3 < KSTEPS * 8) af[(i + 3) & 3] = load_afrag(i + 3);
                const int ks = i >> 3;
                const int rt = i & 7;
                #pragma unroll
                for (int nt = 0; nt < NT; ++nt)
                    acc[rt][nt] = __builtin_amdgcn_wmma_f32_16x16x32_f16(
                        false, af[i & 3], false, bf[ks][nt], (short)0, acc[rt][nt], false, false);
                // enforce [2x DS-read (3 steps ahead), 2x WMMA] interleave
                if (i + 3 < KSTEPS * 8) SGB(0x100, 2);
                SGB(0x8, 2);
            }
        } // kb

        // ---- online softmax partials for this 256-col group ----
        const int   ncol0 = nbase + wave * (NT * 16) + l16;
        const float bb0   = bias[ncol0];
        const float bb1   = bias[ncol0 + 16];
        #pragma unroll
        for (int rt = 0; rt < 8; ++rt) {
            float v0[8], v1[8];
            #pragma unroll
            for (int v = 0; v < 8; ++v) {
                v0[v] = acc[rt][0][v] + bb0;
                v1[v] = acc[rt][1][v] + bb1;
                const int rl = rt * 16 + g * 8 + v;
                const int t  = tl[rl];
                if (t == ncol0)      ptgt[m0 + rl] = v0[v];   // unique writer per row
                if (t == ncol0 + 16) ptgt[m0 + rl] = v1[v];
            }
            float rmax[8], rsum[8];
            #pragma unroll
            for (int v = 0; v < 8; ++v) {
                float mx = fmaxf(v0[v], v1[v]);
                mx = fmaxf(mx, __shfl_xor(mx, 1, 32));
                mx = fmaxf(mx, __shfl_xor(mx, 2, 32));
                mx = fmaxf(mx, __shfl_xor(mx, 4, 32));
                mx = fmaxf(mx, __shfl_xor(mx, 8, 32));
                rmax[v] = mx;
                float e = __expf(v0[v] - mx) + __expf(v1[v] - mx);
                e += __shfl_xor(e, 1, 32);
                e += __shfl_xor(e, 2, 32);
                e += __shfl_xor(e, 4, 32);
                e += __shfl_xor(e, 8, 32);
                rsum[v] = e;
            }
            if (l16 == 0) {
                #pragma unroll
                for (int v = 0; v < 8; ++v) {
                    const int rl = rt * 16 + g * 8 + v;
                    wmx[wave][rl] = rmax[v];
                    wsm[wave][rl] = rsum[v];
                }
            }
        }
        __syncthreads();   // all wave slots written (full fence fine here, cold path)
        if (tid < BM) {
            float m = wmx[0][tid];
            #pragma unroll
            for (int wv = 1; wv < 8; ++wv) m = fmaxf(m, wmx[wv][tid]);
            float s = 0.f;
            #pragma unroll
            for (int wv = 0; wv < 8; ++wv) s += wsm[wv][tid] * __expf(wmx[wv][tid] - m);
            const float nm = fmaxf(run_m, m);
            run_s = run_s * __expf(run_m - nm) + s * __expf(m - nm);
            run_m = nm;
        }
        // next group's first kb-barrier orders slot reuse
    } // ng

    if (tid < BM) {
        pmax[(size_t)(m0 + tid) * NCHUNK + chunk] = run_m;
        psum[(size_t)(m0 + tid) * NCHUNK + chunk] = run_s;
    }
}

// ---------------- per-row LSE merge across the 125 N-chunks ----------------
__launch_bounds__(256)
__global__ void ce_row_kernel(const float* __restrict__ pmax,
                              const float* __restrict__ psum,
                              const float* __restrict__ ptgt,
                              const long long* __restrict__ tgt,
                              float* __restrict__ prow)
{
    const int row = blockIdx.x * blockDim.x + threadIdx.x;
    if (row >= M_ROWS) return;
    float m = -__builtin_inff();
    float s = 0.f;
    for (int c = 0; c < NCHUNK; ++c) {
        const float mi = pmax[(size_t)row * NCHUNK + c];
        const float si = psum[(size_t)row * NCHUNK + c];
        const float nm = fmaxf(m, mi);
        s = s * __expf(m - nm) + si * __expf(mi - nm);
        m = nm;
    }
    const float lse = m + __logf(s);
    const long long t = tgt[row];
    prow[row] = (t != (long long)IGNORE_IDX) ? (lse - ptgt[row]) : 0.f;
}

// ---------------- deterministic final reduction ----------------
__launch_bounds__(256)
__global__ void ce_reduce_kernel(const float* __restrict__ prow,
                                 const long long* __restrict__ tgt,
                                 float* __restrict__ out)
{
    __shared__ float ssum[256];
    __shared__ int   scnt[256];
    const int tid = threadIdx.x;
    float s = 0.f; int c = 0;
    for (int i = tid; i < M_ROWS; i += 256) {
        s += prow[i];
        c += (tgt[i] != (long long)IGNORE_IDX) ? 1 : 0;
    }
    ssum[tid] = s; scnt[tid] = c;
    __syncthreads();
    for (int off = 128; off > 0; off >>= 1) {
        if (tid < off) { ssum[tid] += ssum[tid + off]; scnt[tid] += scnt[tid + off]; }
        __syncthreads();
    }
    if (tid == 0) {
        const int nv = scnt[0] > 1 ? scnt[0] : 1;
        out[0] = ssum[0] / (float)nv;
    }
}

extern "C" void kernel_launch(void* const* d_in, const int* in_sizes, int n_in,
                              void* d_out, int out_size, void* d_ws, size_t ws_size,
                              hipStream_t stream) {
    (void)in_sizes; (void)n_in; (void)out_size; (void)ws_size;
    const float*     x = (const float*)d_in[0];
    const float*     w = (const float*)d_in[1];
    const float*     b = (const float*)d_in[2];
    const long long* t = (const long long*)d_in[3];

    float* ws   = (float*)d_ws;
    float* ptgt = ws;                              // 4096 floats
    float* prow = ws + M_ROWS;                     // 4096 floats
    float* pmax = ws + 2 * M_ROWS;                 // 4096*125 floats
    float* psum = pmax + (size_t)M_ROWS * NCHUNK;  // 4096*125 floats

    dim3 grid(NCHUNK, M_ROWS / BM);                // 125 x 32
    ce_main_kernel<<<grid, 256, 0, stream>>>(x, w, b, t, pmax, psum, ptgt);
    ce_row_kernel<<<dim3(M_ROWS / 256), 256, 0, stream>>>(pmax, psum, ptgt, t, prow);
    ce_reduce_kernel<<<dim3(1), 256, 0, stream>>>(prow, t, (float*)d_out);
}
